// STST_option_17746804867254
// MI455X (gfx1250) — matrix operand order repaired and checked by reading, not compiled, not run on hardware
//
#include <hip/hip_runtime.h>
#include <hip/hip_bf16.h>
#include <math.h>

typedef __attribute__((ext_vector_type(16))) __bf16 v16bf;
typedef __attribute__((ext_vector_type(8)))  float  v8f;

#define BN_INV 0.999995000037499687f

__device__ __forceinline__ unsigned short f2bf(float f) {
  union { float f; unsigned u; } v; v.f = f;
  unsigned r = v.u + 0x7fffu + ((v.u >> 16) & 1u);
  return (unsigned short)(r >> 16);
}
__device__ __forceinline__ float bf2f(unsigned short s) {
  union { unsigned u; float f; } v; v.u = ((unsigned)s) << 16;
  return v.f;
}
__device__ __forceinline__ __bf16 us2bf(unsigned short s) {
  __bf16 b; __builtin_memcpy(&b, &s, 2); return b;
}

// ---------------------------------------------------------------------------
// f32 -> bf16 conversion
// ---------------------------------------------------------------------------
__global__ __launch_bounds__(256) void cvt_bf16(const float* __restrict__ src,
                                                unsigned short* __restrict__ dst,
                                                int n) {
  int i = blockIdx.x * 256 + threadIdx.x;
  if (i < n) dst[i] = f2bf(src[i]);
}

// ---------------------------------------------------------------------------
// Generic WMMA bf16 GEMM, double-buffered LDS, async A-tile staging.
//   out[n][o][p] = act( Σ_k W[o][k] * B(k,p) )
// bsrc: 0 = X row-major [n][K][P]
//       1 = temporal einsum gather: W per (n,s) block (A2), X = z_in[n][c][t][v],
//           row k = t, col p = c*25+v
//       2 = TCN im2col: row kk -> (c=kk/7, dk=kk%7), value X[n][c][t+dk-3][v]
// ep:   0 = outB = bf16(acc + bias)
//       1 = val = lrelu(resid + (acc+bias)*g*BN_INV + be); outF & outB
//       2 = z-scatter: outB[n][s*128+c][o*25+v] = bf16(acc)
//       3 = like 1 but outF only (final output)
// ---------------------------------------------------------------------------
// LDS layout (u16 elements), pitch 40 so every 16B granule is aligned:
//   A tile buf b: [b*2560 + m*40 + k],  m<64, k<32
//   B tile buf b: [5120 + b*2560 + p*40 + k]  (transposed: p rows, k contiguous)
#define AIDX(b, r, c) ((b) * 2560 + (r) * 40 + (c))
#define BIDX(b, r, c) (5120 + (b) * 2560 + (r) * 40 + (c))

__global__ __launch_bounds__(256) void gemm_wmma(
    const unsigned short* __restrict__ Wt,
    const unsigned short* __restrict__ Xb,
    const float* __restrict__ bias,
    const float* __restrict__ gamma,
    const float* __restrict__ beta,
    const float* __restrict__ resid,
    float* __restrict__ outF,
    unsigned short* __restrict__ outB,
    int O, int K, int P, int bsrc, int ep) {
  int tid = threadIdx.x;
  int z = blockIdx.z;
  int n, s4 = 0;
  const unsigned short* Wp;
  const unsigned short* Xn;
  if (bsrc == 1) {
    n = z >> 2; s4 = z & 3;
    Wp = Wt + (size_t)z * O * K;
    Xn = Xb + (size_t)n * 128 * 3200;
  } else if (bsrc == 2) {
    n = z; Wp = Wt;
    Xn = Xb + (size_t)n * 128 * 3200;
  } else {
    n = z; Wp = Wt;
    Xn = Xb + (size_t)n * K * P;
  }
  int obase = blockIdx.y * 64;
  int pbase = blockIdx.x * 64;

  __shared__ unsigned short SH[10240];   // 2xA (5120) + 2xB (5120) u16 = 20 KB

  int lane = tid & 31, wv = tid >> 5;
  int wm = wv & 3, wn = wv >> 2;         // 4 M-tiles x 2 N-tiles of waves
  int h = lane >> 4, lm = lane & 15;
  int ra = tid >> 2, ch = tid & 3;       // A loader: row, 16B chunk
  int kl = tid & 31, nb = tid >> 5;      // B loader: k row, 8-col group

  v8f acc0 = {0.f,0.f,0.f,0.f,0.f,0.f,0.f,0.f};
  v8f acc1 = {0.f,0.f,0.f,0.f,0.f,0.f,0.f,0.f};

  const int nk = K >> 5;

  // --- A tile: async DMA global->LDS, 16B per lane, 256 lanes = 64x32 tile ---
  auto stage_a = [&](int k0, int buf) {
    unsigned lds = (unsigned)(unsigned long long)(size_t)&SH[AIDX(buf, ra, ch * 8)];
    unsigned long long g =
        (unsigned long long)(size_t)(Wp + (size_t)(obase + ra) * K + k0 + ch * 8);
    asm volatile("global_load_async_to_lds_b128 %0, %1, off"
                 :: "v"(lds), "v"(g) : "memory");
  };
  // --- B tile: load to regs (issued before compute), scatter to LDS after ---
  auto load_b = [&](int k0, uint4& bv, unsigned short* br) {
    if (bsrc == 0) {
      bv = *reinterpret_cast<const uint4*>(Xn + (size_t)(k0 + kl) * P + pbase + nb * 8);
    } else if (bsrc == 1) {
#pragma unroll
      for (int i = 0; i < 8; ++i) {
        int p = pbase + nb * 8 + i;
        int c = p / 25, v = p - c * 25;
        br[i] = Xn[(size_t)c * 3200 + (size_t)(k0 + kl) * 25 + v];
      }
    } else {
      int kk = k0 + kl, c = kk / 7, dk = kk - c * 7;
#pragma unroll
      for (int i = 0; i < 8; ++i) {
        int p = pbase + nb * 8 + i;
        int tt = p / 25, v = p - tt * 25;
        int ts = tt + dk - 3;
        br[i] = (ts >= 0 && ts < 128) ? Xn[(size_t)c * 3200 + ts * 25 + v]
                                      : (unsigned short)0;
      }
    }
  };
  auto store_b = [&](int buf, const uint4& bv, const unsigned short* br) {
    if (bsrc == 0) {
      union { uint4 v; unsigned short h[8]; } u; u.v = bv;
#pragma unroll
      for (int i = 0; i < 8; ++i) SH[BIDX(buf, nb * 8 + i, kl)] = u.h[i];
    } else {
#pragma unroll
      for (int i = 0; i < 8; ++i) SH[BIDX(buf, nb * 8 + i, kl)] = br[i];
    }
  };

  // prologue: stage tile 0
  {
    uint4 bv = {0, 0, 0, 0}; unsigned short br[8];
    stage_a(0, 0);
    load_b(0, bv, br);
    store_b(0, bv, br);
  }
  asm volatile("s_wait_asynccnt 0x0" ::: "memory");
  __syncthreads();

  for (int t = 0; t < nk; ++t) {
    int cur = t & 1, nxt = cur ^ 1;
    bool has = (t + 1) < nk;
    uint4 bv = {0, 0, 0, 0}; unsigned short br[8];
    if (has) {              // issue next tile's DMA + global loads before compute
      stage_a((t + 1) << 5, nxt);
      load_b((t + 1) << 5, bv, br);
    }

    // A fragment (16x32): lane L: M=L&15; elem e: K=(L>>4)*16 + e
    v16bf af;
#pragma unroll
    for (int e = 0; e < 16; ++e)
      af[e] = us2bf(SH[AIDX(cur, wm * 16 + lm, (h << 4) + e)]);
    // B fragments (32x16): lane L: N=L&15; elem e: K=(L>>4)*16 + e
    v16bf b0, b1;
#pragma unroll
    for (int e = 0; e < 16; ++e) {
      b0[e] = us2bf(SH[BIDX(cur, wn * 32 + lm, (h << 4) + e)]);
      b1[e] = us2bf(SH[BIDX(cur, wn * 32 + 16 + lm, (h << 4) + e)]);
    }
    acc0 = __builtin_amdgcn_wmma_f32_16x16x32_bf16(false, af, false, b0, (short)0, acc0, false, false);
    acc1 = __builtin_amdgcn_wmma_f32_16x16x32_bf16(false, af, false, b1, (short)0, acc1, false, false);

    if (has) store_b(nxt, bv, br);
    asm volatile("s_wait_asynccnt 0x0" ::: "memory");
    __syncthreads();
  }

  // epilogue: D elem r -> M = r + 8*h, N = lane&15
  size_t nOP = (size_t)n * O * P;
#pragma unroll
  for (int j = 0; j < 2; ++j) {
    v8f a = j ? acc1 : acc0;
    int p = pbase + wn * 32 + j * 16 + lm;
#pragma unroll
    for (int r = 0; r < 8; ++r) {
      int o = obase + wm * 16 + h * 8 + r;
      float accv = a[r];
      if (ep == 0) {
        outB[nOP + (size_t)o * P + p] = f2bf(accv + bias[o]);
      } else if (ep == 2) {
        int c = p / 25, vv = p - c * 25;
        outB[(size_t)n * 512 * 3200 + (size_t)(s4 * 128 + c) * 3200 + o * 25 + vv] = f2bf(accv);
      } else {
        float tq = (accv + bias[o]) * (gamma[o] * BN_INV) + beta[o];
        float rv = resid[nOP + (size_t)o * P + p] + tq;
        float val = rv > 0.f ? rv : 0.1f * rv;
        outF[nOP + (size_t)o * P + p] = val;
        if (ep == 1) outB[nOP + (size_t)o * P + p] = f2bf(val);
      }
    }
  }
}

// ---------------------------------------------------------------------------
// Spatial attention: att[n,s,t,u,v] = att0[s,u,v] + tanh(q.k / 32) * alpha[s]
// grid (T, HS, N), 256 threads
// ---------------------------------------------------------------------------
__global__ __launch_bounds__(256) void att_spatial(
    const unsigned short* __restrict__ qk, const float* __restrict__ att0,
    const float* __restrict__ alphas, float* __restrict__ att) {
  int t = blockIdx.x, s = blockIdx.y, n = blockIdx.z;
  __shared__ float qs[32][25], ks[32][25];
  const unsigned short* qb = qk + ((size_t)n * 192 + s * 32) * 3200 + t * 25;
  const unsigned short* kb = qk + ((size_t)n * 192 + 96 + s * 32) * 3200 + t * 25;
  for (int i = threadIdx.x; i < 800; i += 256) {
    int ci = i / 25, u = i - ci * 25;
    qs[ci][u] = bf2f(qb[(size_t)ci * 3200 + u]);
    ks[ci][u] = bf2f(kb[(size_t)ci * 3200 + u]);
  }
  __syncthreads();
  float alpha = alphas[s];
  for (int i = threadIdx.x; i < 625; i += 256) {
    int u = i / 25, v = i - u * 25;
    float acc = 0.f;
#pragma unroll
    for (int ci = 0; ci < 32; ++ci) acc += qs[ci][u] * ks[ci][v];
    att[(((size_t)n * 3 + s) * 128 + t) * 625 + i] =
        att0[s * 625 + u * 25 + v] + tanhf(acc * (1.f / 32.f)) * alpha;
  }
}

// ---------------------------------------------------------------------------
// y[n, s*128+c, t, v] = Σ_u x[n,c,t,u] * att[n,s,t,u,v]; grid (T, N)
// ---------------------------------------------------------------------------
__global__ __launch_bounds__(256) void spatial_apply(
    const unsigned short* __restrict__ xbf, const float* __restrict__ att,
    unsigned short* __restrict__ ys) {
  int t = blockIdx.x, n = blockIdx.y;
  __shared__ float xs[128][25];
  __shared__ float as_[3][25][25];
  const unsigned short* xb = xbf + (size_t)n * 128 * 3200 + t * 25;
  for (int i = threadIdx.x; i < 3200; i += 256) {
    int c = i / 25, u = i - c * 25;
    xs[c][u] = bf2f(xb[(size_t)c * 3200 + u]);
  }
  for (int i = threadIdx.x; i < 1875; i += 256) {
    int s = i / 625, r = i - s * 625;
    as_[s][r / 25][r % 25] = att[(((size_t)n * 3 + s) * 128 + t) * 625 + r];
  }
  __syncthreads();
  for (int i = threadIdx.x; i < 9600; i += 256) {
    int sc = i / 25, v = i - sc * 25;
    int s = sc >> 7, c = sc & 127;
    float acc = 0.f;
#pragma unroll
    for (int u = 0; u < 25; ++u) acc += xs[c][u] * as_[s][u][v];
    ys[((size_t)n * 384 + sc) * 3200 + t * 25 + v] = f2bf(acc);
  }
}

// ---------------------------------------------------------------------------
// xbar[n][c][t] = mean_v s_out[n][c][t][v]  (bf16 out, feeds qkt GEMM)
// ---------------------------------------------------------------------------
__global__ __launch_bounds__(256) void mean_v(const float* __restrict__ sout,
                                              unsigned short* __restrict__ xbar) {
  int idx = blockIdx.x * 256 + threadIdx.x;
  if (idx >= 32 * 128 * 128) return;
  const float* p = sout + (size_t)idx * 25;
  float a = 0.f;
#pragma unroll
  for (int v = 0; v < 25; ++v) a += p[v];
  xbar[idx] = f2bf(a * (1.f / 25.f));
}

// ---------------------------------------------------------------------------
// Temporal attention, stored transposed for GEMM:
//   a2[n][s][u][t] = tanh(qt.kt/32) * alpha * mask(t,u)
// grid 128 (= n*4+s), 256 threads
// ---------------------------------------------------------------------------
__global__ __launch_bounds__(256) void att_temporal(
    const unsigned short* __restrict__ qkt, const float* __restrict__ af,
    const float* __restrict__ ab, unsigned short* __restrict__ a2) {
  int s = blockIdx.x & 3, n = blockIdx.x >> 2;
  __shared__ float qs[32][128], ks[32][128];
  const unsigned short* qb = qkt + ((size_t)n * 256 + s * 32) * 128;
  const unsigned short* kb = qkt + ((size_t)n * 256 + 128 + s * 32) * 128;
  for (int i = threadIdx.x; i < 4096; i += 256) {
    int ci = i >> 7, t = i & 127;
    qs[ci][t] = bf2f(qb[ci * 128 + t]);
    ks[ci][t] = bf2f(kb[ci * 128 + t]);
  }
  __syncthreads();
  float alpha = (s < 2) ? af[s] : ab[s - 2];
  for (int i = threadIdx.x; i < 16384; i += 256) {
    int t = i >> 7, u = i & 127;
    float m = (s < 2) ? (t >= u ? 1.f : 0.f) : (u >= t ? 1.f : 0.f);
    float acc = 0.f;
#pragma unroll
    for (int ci = 0; ci < 32; ++ci) acc += qs[ci][t] * ks[ci][u];
    a2[(((size_t)n * 4 + s) * 128 + u) * 128 + t] = f2bf(tanhf(acc * (1.f / 32.f)) * alpha * m);
  }
}

// ---------------------------------------------------------------------------
extern "C" void kernel_launch(void* const* d_in, const int* in_sizes, int n_in,
                              void* d_out, int out_size, void* d_ws, size_t ws_size,
                              hipStream_t stream) {
  (void)in_sizes; (void)n_in; (void)out_size; (void)ws_size;
  const float* x       = (const float*)d_in[0];
  const float* w_in_s  = (const float*)d_in[1];
  const float* b_in_s  = (const float*)d_in[2];
  const float* att0s   = (const float*)d_in[3];
  const float* alphas  = (const float*)d_in[4];
  const float* w_out_s = (const float*)d_in[5];
  const float* b_out_s = (const float*)d_in[6];
  const float* g_out_s = (const float*)d_in[7];
  const float* be_out_s= (const float*)d_in[8];
  const float* w_ff_s  = (const float*)d_in[9];
  const float* b_ff_s  = (const float*)d_in[10];
  const float* g_ff_s  = (const float*)d_in[11];
  const float* be_ff_s = (const float*)d_in[12];
  const float* w_in_t  = (const float*)d_in[13];
  const float* b_in_t  = (const float*)d_in[14];
  const float* at_f    = (const float*)d_in[15];
  const float* at_b    = (const float*)d_in[16];
  const float* w_out_t = (const float*)d_in[17];
  const float* b_out_t = (const float*)d_in[18];
  const float* g_out_t = (const float*)d_in[19];
  const float* be_out_t= (const float*)d_in[20];
  const float* w_ff_t  = (const float*)d_in[21];
  const float* b_ff_t  = (const float*)d_in[22];
  const float* g_ff_t  = (const float*)d_in[23];
  const float* be_ff_t = (const float*)d_in[24];
  const float* w_tcn   = (const float*)d_in[25];
  const float* b_tcn   = (const float*)d_in[26];
  const float* g_tcn   = (const float*)d_in[27];
  const float* be_tcn  = (const float*)d_in[28];

  char* ws = (char*)d_ws;
  unsigned short* XBF = (unsigned short*)(ws + 0);           // x bf16 (26.2 MB)
  unsigned short* WBF = (unsigned short*)(ws + 26214400);    // weights bf16 (1 MB)
  unsigned short* BIG = (unsigned short*)(ws + 27262976);    // qk -> ys -> z (104.9 MB)
  char* SM = ws + 132120576;                                 // 32 MB scratch region
  float* ATT           = (float*)SM;                         // spatial att (30.7 MB)
  unsigned short* XBAR = (unsigned short*)SM;                // reused after ATT dead
  unsigned short* QKT  = (unsigned short*)(SM + 1048576);
  unsigned short* A2   = (unsigned short*)(SM + 3145728);
  float*          R1   = (float*)(ws + 165675008);           // stage-1 f32
  unsigned short* R1B  = (unsigned short*)(ws + 218103808);
  float*          R2   = (float*)(ws + 244318208);           // s_out f32
  unsigned short* R2B  = (unsigned short*)(ws + 296747008);
  float*          R3   = (float*)(ws + 322961408);           // t_out f32
  unsigned short* R3B  = (unsigned short*)(ws + 375390208);

  unsigned short* Wis = WBF;            // 192x128
  unsigned short* Wos = WBF + 24576;    // 128x384
  unsigned short* Wfs = WBF + 73728;    // 128x128
  unsigned short* Wit = WBF + 90112;    // 256x128
  unsigned short* Wot = WBF + 122880;   // 128x512
  unsigned short* Wft = WBF + 188416;   // 128x128
  unsigned short* Wtc = WBF + 204800;   // 128x896

  dim3 blk(256);
#define NB(n) (((n) + 255) / 256)
  cvt_bf16<<<NB(13107200), blk, 0, stream>>>(x, XBF, 13107200);
  cvt_bf16<<<NB(24576),  blk, 0, stream>>>(w_in_s,  Wis, 24576);
  cvt_bf16<<<NB(49152),  blk, 0, stream>>>(w_out_s, Wos, 49152);
  cvt_bf16<<<NB(16384),  blk, 0, stream>>>(w_ff_s,  Wfs, 16384);
  cvt_bf16<<<NB(32768),  blk, 0, stream>>>(w_in_t,  Wit, 32768);
  cvt_bf16<<<NB(65536),  blk, 0, stream>>>(w_out_t, Wot, 65536);
  cvt_bf16<<<NB(16384),  blk, 0, stream>>>(w_ff_t,  Wft, 16384);
  cvt_bf16<<<NB(114688), blk, 0, stream>>>(w_tcn,   Wtc, 114688);

  // qk = W_in_s * x + b  (192x128x3200, per n)
  gemm_wmma<<<dim3(50, 3, 32), blk, 0, stream>>>(Wis, XBF, b_in_s, nullptr, nullptr,
      nullptr, nullptr, BIG, 192, 128, 3200, 0, 0);
  // spatial attention scores
  att_spatial<<<dim3(128, 3, 32), blk, 0, stream>>>(BIG, att0s, alphas, ATT);
  // y = x @ att  -> BIG (384 ch, overwrites qk)
  spatial_apply<<<dim3(128, 32), blk, 0, stream>>>(XBF, ATT, BIG);
  // s1 = lrelu(x + bn(W_out_s*y + b))
  gemm_wmma<<<dim3(50, 2, 32), blk, 0, stream>>>(Wos, BIG, b_out_s, g_out_s, be_out_s,
      x, R1, R1B, 128, 384, 3200, 0, 1);
  // s_out = lrelu(x + bn(W_ff_s*s1 + b))
  gemm_wmma<<<dim3(50, 2, 32), blk, 0, stream>>>(Wfs, R1B, b_ff_s, g_ff_s, be_ff_s,
      x, R2, R2B, 128, 128, 3200, 0, 1);
  // temporal: mean over V, small GEMM, masked attention
  mean_v<<<2048, blk, 0, stream>>>(R2, XBAR);
  gemm_wmma<<<dim3(2, 4, 32), blk, 0, stream>>>(Wit, XBAR, b_in_t, nullptr, nullptr,
      nullptr, nullptr, QKT, 256, 128, 128, 0, 0);
  att_temporal<<<128, blk, 0, stream>>>(QKT, at_f, at_b, A2);
  // z[n][s*128+c][u][v] = Σ_t a2[n,s,u,t] * s_out[n,c,t,v]  -> BIG (512 ch)
  gemm_wmma<<<dim3(50, 2, 128), blk, 0, stream>>>(A2, R2B, nullptr, nullptr, nullptr,
      nullptr, nullptr, BIG, 128, 128, 3200, 1, 2);
  // t1 = lrelu(s_out + bn(W_out_t*z + b))
  gemm_wmma<<<dim3(50, 2, 32), blk, 0, stream>>>(Wot, BIG, b_out_t, g_out_t, be_out_t,
      R2, R1, R1B, 128, 512, 3200, 0, 1);
  // t_out = lrelu(s_out + bn(W_ff_t*t1 + b))
  gemm_wmma<<<dim3(50, 2, 32), blk, 0, stream>>>(Wft, R1B, b_ff_t, g_ff_t, be_ff_t,
      R2, R3, R3B, 128, 128, 3200, 0, 1);
  // out = lrelu(t_out + bn(tcn(t_out) + b))   (im2col K = 128*7 = 896)
  gemm_wmma<<<dim3(50, 2, 32), blk, 0, stream>>>(Wtc, R3B, b_tcn, g_tcn, be_tcn,
      R3, (float*)d_out, nullptr, 128, 896, 3200, 2, 3);
#undef NB
}
